// SimpleCorrector_89481348645712
// MI455X (gfx1250) — matrix operand order repaired and checked by reading, not compiled, not run on hardware
//
#include <hip/hip_runtime.h>

typedef __attribute__((ext_vector_type(4)))  _Float16 v4h;
typedef __attribute__((ext_vector_type(8)))  _Float16 v8h;
typedef __attribute__((ext_vector_type(16))) _Float16 v16h;
typedef __attribute__((ext_vector_type(8)))  float    v8f;

#define NNODES 100000
#define NEDGES 1600000
#define FIN    64
#define HID    128
#define KPAD0  96     // msg layer0 K padded: 71 -> 96 (3 k-steps of 32)
#define KC0    192    // corrector layer0 K: 64 + 128 (6 k-steps of 32)

union V16U { v16h v; v8h h[2]; };

// A fragment (16x32 f16) from LDS, row-major [16][strideH halves].
__device__ __forceinline__ v16h lds_loadA(const _Float16* base, int strideH, int lane, int k0) {
  const int row = lane & 15;
  const int hi  = lane >> 4;
  const _Float16* p = base + row * strideH + k0 + hi * 8;
  V16U u;
  u.h[0] = *(const v8h*)(p);
  u.h[1] = *(const v8h*)(p + 16);
  return u.v;
}

// B fragment (32x16 f16) from transposed weights wT[n*strideK + k].
__device__ __forceinline__ v16h lds_loadB(const _Float16* wT, int strideK, int lane, int k0) {
  const int col = lane & 15;
  const int hi  = lane >> 4;
  const _Float16* p = wT + col * strideK + k0 + hi * 8;
  V16U u;
  u.h[0] = *(const v8h*)(p);
  u.h[1] = *(const v8h*)(p + 16);
  return u.v;
}

__device__ __forceinline__ v8f wmma16(v16h a, v16h b, v8f c) {
  return __builtin_amdgcn_wmma_f32_16x16x32_f16(false, a, false, b, (short)0, c, false, false);
}

#if defined(__gfx1250__) && __has_builtin(__builtin_amdgcn_tensor_load_to_lds)
#define EDGE_TDM 1
typedef __attribute__((ext_vector_type(4))) unsigned int v4u;
typedef __attribute__((ext_vector_type(8))) int v8i_;
typedef __attribute__((ext_vector_type(4))) int v4i_;

// TDM: copy `ndw` dwords from global to LDS[lds_off] as one (ndw x 1) 4B tile.
// D# layout per CDNA5 ISA ch.8: group0 = {count, lds_addr, global_addr, type=2},
// group1 = {data_size=4B, tensor_dim0=ndw, tensor_dim1=1, tile_dim0=ndw, tile_dim1=1,
//           tensor_dim0_stride=ndw}; remaining groups unused (<=2D tensor).
// This toolchain exposes the 6-arg builtin: (v4u, v8i, v4i, v4i, v8i, cpol).
__device__ __forceinline__ void tdm_load_dwords(unsigned lds_off, const float* gsrc, unsigned ndw) {
  unsigned long long ga = (unsigned long long)gsrc;
  v4u g0;
  g0[0] = 1u;                                                    // count=1, user-mode
  g0[1] = lds_off;                                               // lds_addr
  g0[2] = (unsigned)ga;                                          // global_addr[31:0]
  g0[3] = (unsigned)((ga >> 32) & 0x01FFFFFFull) | (2u << 30);   // addr[56:32] | type=2
  v8i_ g1;
  g1[0] = (int)(2u << 16);                      // workgroup_mask=0, data_size=4B
  g1[1] = (int)((ndw & 0xFFFFu) << 16);         // tensor_dim0[15:0] @ bits[63:48]
  g1[2] = (int)((ndw >> 16) | (1u << 16));      // tensor_dim0[31:16], tensor_dim1=1
  g1[3] = (int)((ndw & 0xFFFFu) << 16);         // tile_dim0 @ bits[127:112]
  g1[4] = 1;                                    // tile_dim1 = 1
  g1[5] = (int)ndw;                             // tensor_dim0_stride[31:0]
  g1[6] = 0;
  g1[7] = 0;
  v4i_ gz4 = {0, 0, 0, 0};
  v8i_ gz8 = {0, 0, 0, 0, 0, 0, 0, 0};
  __builtin_amdgcn_tensor_load_to_lds(g0, g1, gz4, gz4, gz8, 0);
}
#endif

__global__ void zero_kernel(float* __restrict__ p, int n4) {
  int i = blockIdx.x * blockDim.x + threadIdx.x;
  float4 z = {0.f, 0.f, 0.f, 0.f};
  for (; i < n4; i += gridDim.x * blockDim.x) ((float4*)p)[i] = z;
}

// ---------------- edge message MLP + scatter-add (M=32 edges per wave) ----------------
__global__ void __launch_bounds__(256) edge_msg_kernel(
    const float* __restrict__ x, const long long* __restrict__ eidx,
    const float* __restrict__ W0, const float* __restrict__ b0,
    const float* __restrict__ W1, const float* __restrict__ b1,
    const float* __restrict__ W2, const float* __restrict__ b2,
    float* __restrict__ agg)
{
  __shared__ _Float16 sW0[HID * KPAD0];   // transposed [n][k], k padded to 96 (24KB)
  __shared__ _Float16 sW1[HID * HID];     // transposed [n][k] (32KB)
  __shared__ _Float16 sW2[HID * HID];     // (32KB)
  __shared__ float sB0[HID], sB1[HID], sB2[HID];
  __shared__ _Float16 sA[8][32 * HID];    // per-wave 32-row staging (64KB total)
  __shared__ int sRow[8][32];

  const int tid = threadIdx.x;
#ifdef EDGE_TDM
  // Stage raw f32 weights into the (pre-loop idle) sA region via the Tensor Data
  // Mover, then convert/transpose to f16 in LDS.  One wave issues each DMA.
  float* scratchF = (float*)&sA[0][0];    // 64KB >= 128*128*4
  const unsigned lds_off = (unsigned)(unsigned long long)scratchF;
  if (tid < 32) {
    tdm_load_dwords(lds_off, W0, 71 * HID);
    __builtin_amdgcn_s_wait_tensorcnt(0);
  }
  __syncthreads();
  for (int i = tid; i < HID * KPAD0; i += 256) {
    int n = i / KPAD0, k = i - n * KPAD0;
    sW0[i] = (k < 71) ? (_Float16)scratchF[k * HID + n] : (_Float16)0.0f;
  }
  __syncthreads();
  if (tid < 32) {
    tdm_load_dwords(lds_off, W1, HID * HID);
    __builtin_amdgcn_s_wait_tensorcnt(0);
  }
  __syncthreads();
  for (int i = tid; i < HID * HID; i += 256) {
    int n = i >> 7, k = i & 127;
    sW1[i] = (_Float16)scratchF[k * HID + n];
  }
  __syncthreads();
  if (tid < 32) {
    tdm_load_dwords(lds_off, W2, HID * HID);
    __builtin_amdgcn_s_wait_tensorcnt(0);
  }
  __syncthreads();
  for (int i = tid; i < HID * HID; i += 256) {
    int n = i >> 7, k = i & 127;
    sW2[i] = (_Float16)scratchF[k * HID + n];
  }
#else
  for (int i = tid; i < HID * KPAD0; i += 256) {
    int n = i / KPAD0, k = i - n * KPAD0;
    sW0[i] = (k < 71) ? (_Float16)W0[k * HID + n] : (_Float16)0.0f;
  }
  for (int i = tid; i < HID * HID; i += 256) {
    int n = i >> 7, k = i & 127;
    sW1[i] = (_Float16)W1[k * HID + n];
    sW2[i] = (_Float16)W2[k * HID + n];
  }
#endif
  if (tid < HID) { sB0[tid] = b0[tid]; sB1[tid] = b1[tid]; sB2[tid] = b2[tid]; }
  __syncthreads();

  const int lane = tid & 31;
  const int wave = tid >> 5;
  const int hi   = lane >> 4;
  const int col0 = lane & 15;
  _Float16* A = sA[wave];
  const int nTiles = (NEDGES + 31) >> 5;   // 32 edges per wave-iteration

  for (int tile = blockIdx.x * 8 + wave; tile < nTiles; tile += gridDim.x * 8) {
    const int e0 = tile << 5;
    int e = e0 + lane; if (e >= NEDGES) e = NEDGES - 1;
    __builtin_prefetch(eidx + (long long)(tile + gridDim.x * 8) * 32 + lane, 0, 1);
    const long long r = eidx[e];
    const long long c = eidx[(long long)NEDGES + e];
    sRow[wave][lane] = (int)r;

    // each lane owns one edge: gather 64 feats + geometry, no divergence
    {
      const float* xr = x + r * FIN;
      _Float16* arow = A + lane * HID;
      #pragma unroll
      for (int f = 0; f < FIN; f += 4) {
        float4 v = *(const float4*)(xr + f);
        v4h h4 = { (_Float16)v.x, (_Float16)v.y, (_Float16)v.z, (_Float16)v.w };
        *(v4h*)(arow + f) = h4;
      }
      const float* xj = x + c * FIN;
      float rx = xj[0] - xr[0], ry = xj[1] - xr[1], rz = xj[2] - xr[2];
      float d  = sqrtf(rx*rx + ry*ry + rz*rz) + 1e-12f;
      float iv = 1.0f / d;
      arow[64] = (_Float16)rx; arow[65] = (_Float16)ry; arow[66] = (_Float16)rz;
      arow[67] = (_Float16)d;
      arow[68] = (_Float16)(rx*iv); arow[69] = (_Float16)(ry*iv); arow[70] = (_Float16)(rz*iv);
      arow[71] = (_Float16)0.0f;
      v4h z4 = { (_Float16)0.f, (_Float16)0.f, (_Float16)0.f, (_Float16)0.f };
      #pragma unroll
      for (int k = 72; k < KPAD0; k += 4) *(v4h*)(arow + k) = z4;
    }

    // ---- layer 0: (2x16)x96 @ 96x128 ; each B fragment feeds two WMMAs ----
    {
      v16h a[2][3];
      #pragma unroll
      for (int t = 0; t < 2; t++) {
        const _Float16* At = A + t * 16 * HID;
        a[t][0] = lds_loadA(At, HID, lane, 0);
        a[t][1] = lds_loadA(At, HID, lane, 32);
        a[t][2] = lds_loadA(At, HID, lane, 64);
      }
      #pragma unroll
      for (int nt = 0; nt < 8; nt++) {
        float bv = sB0[nt * 16 + col0];
        v8f acc0 = { bv, bv, bv, bv, bv, bv, bv, bv };
        v8f acc1 = acc0;
        const _Float16* wb = sW0 + nt * 16 * KPAD0;
        #pragma unroll
        for (int ks = 0; ks < 3; ks++) {
          v16h b = lds_loadB(wb, KPAD0, lane, ks * 32);
          acc0 = wmma16(a[0][ks], b, acc0);
          acc1 = wmma16(a[1][ks], b, acc1);
        }
        #pragma unroll
        for (int j = 0; j < 8; j++) {
          float v0 = acc0[j]; v0 = v0 > 0.0f ? v0 : 0.0f;
          float v1 = acc1[j]; v1 = v1 > 0.0f ? v1 : 0.0f;
          A[(j + hi * 8) * HID + nt * 16 + col0]        = (_Float16)v0;
          A[(16 + j + hi * 8) * HID + nt * 16 + col0]   = (_Float16)v1;
        }
      }
    }
    // ---- layer 1: (2x16)x128 @ 128x128 ----
    {
      v16h a[2][4];
      #pragma unroll
      for (int t = 0; t < 2; t++) {
        const _Float16* At = A + t * 16 * HID;
        #pragma unroll
        for (int ks = 0; ks < 4; ks++) a[t][ks] = lds_loadA(At, HID, lane, ks * 32);
      }
      #pragma unroll
      for (int nt = 0; nt < 8; nt++) {
        float bv = sB1[nt * 16 + col0];
        v8f acc0 = { bv, bv, bv, bv, bv, bv, bv, bv };
        v8f acc1 = acc0;
        const _Float16* wb = sW1 + nt * 16 * HID;
        #pragma unroll
        for (int ks = 0; ks < 4; ks++) {
          v16h b = lds_loadB(wb, HID, lane, ks * 32);
          acc0 = wmma16(a[0][ks], b, acc0);
          acc1 = wmma16(a[1][ks], b, acc1);
        }
        #pragma unroll
        for (int j = 0; j < 8; j++) {
          float v0 = acc0[j]; v0 = v0 > 0.0f ? v0 : 0.0f;
          float v1 = acc1[j]; v1 = v1 > 0.0f ? v1 : 0.0f;
          A[(j + hi * 8) * HID + nt * 16 + col0]        = (_Float16)v0;
          A[(16 + j + hi * 8) * HID + nt * 16 + col0]   = (_Float16)v1;
        }
      }
    }
    // ---- layer 2: (2x16)x128 @ 128x128, relu, scatter-add ----
    {
      v16h a[2][4];
      #pragma unroll
      for (int t = 0; t < 2; t++) {
        const _Float16* At = A + t * 16 * HID;
        #pragma unroll
        for (int ks = 0; ks < 4; ks++) a[t][ks] = lds_loadA(At, HID, lane, ks * 32);
      }
      #pragma unroll
      for (int nt = 0; nt < 8; nt++) {
        float bv = sB2[nt * 16 + col0];
        v8f acc0 = { bv, bv, bv, bv, bv, bv, bv, bv };
        v8f acc1 = acc0;
        const _Float16* wb = sW2 + nt * 16 * HID;
        #pragma unroll
        for (int ks = 0; ks < 4; ks++) {
          v16h b = lds_loadB(wb, HID, lane, ks * 32);
          acc0 = wmma16(a[0][ks], b, acc0);
          acc1 = wmma16(a[1][ks], b, acc1);
        }
        #pragma unroll
        for (int j = 0; j < 8; j++) {
          int M0 = j + hi * 8;
          int M1 = 16 + j + hi * 8;
          if (e0 + M0 < NEDGES) {
            float v = acc0[j]; v = v > 0.0f ? v : 0.0f;
            atomicAdd(agg + (long long)sRow[wave][M0] * HID + nt * 16 + col0, v);
          }
          if (e0 + M1 < NEDGES) {
            float v = acc1[j]; v = v > 0.0f ? v : 0.0f;
            atomicAdd(agg + (long long)sRow[wave][M1] * HID + nt * 16 + col0, v);
          }
        }
      }
    }
  }
}

// ---------------- per-node corrector MLP ----------------
__global__ void __launch_bounds__(128) corrector_kernel(
    const float* __restrict__ x, const float* __restrict__ agg,
    const float* __restrict__ W0, const float* __restrict__ b0,
    const float* __restrict__ W1, const float* __restrict__ b1,
    const float* __restrict__ W2, const float* __restrict__ b2,
    const float* __restrict__ W3, const float* __restrict__ b3,
    float* __restrict__ out)
{
  __shared__ _Float16 sW0[HID * KC0];     // transposed [n][k], k<192 (48KB)
  __shared__ _Float16 sW1[HID * HID];
  __shared__ _Float16 sW2[HID * HID];
  __shared__ _Float16 sW3[16 * HID];      // cols 3..15 zero
  __shared__ float sB0[HID], sB1[HID], sB2[HID], sB3[16];
  __shared__ _Float16 sA[4][16 * KC0];    // per-wave staging (stride 192)

  const int tid = threadIdx.x;
  for (int i = tid; i < HID * KC0; i += 128) {
    int n = i / KC0, k = i - n * KC0;
    sW0[i] = (_Float16)W0[k * HID + n];
  }
  for (int i = tid; i < HID * HID; i += 128) {
    int n = i >> 7, k = i & 127;
    sW1[i] = (_Float16)W1[k * HID + n];
    sW2[i] = (_Float16)W2[k * HID + n];
  }
  for (int i = tid; i < 16 * HID; i += 128) {
    int n = i >> 7, k = i & 127;
    sW3[i] = (n < 3) ? (_Float16)W3[k * 3 + n] : (_Float16)0.0f;
  }
  if (tid < HID) { sB0[tid] = b0[tid]; sB1[tid] = b1[tid]; sB2[tid] = b2[tid]; }
  if (tid < 16) sB3[tid] = (tid < 3) ? b3[tid] : 0.0f;
  __syncthreads();

  const int lane = tid & 31;
  const int wave = tid >> 5;
  const int ei   = lane & 15;
  const int hi   = lane >> 4;
  const int col0 = lane & 15;
  _Float16* A = sA[wave];
  const int nTiles = (NNODES + 15) >> 4;

  for (int tile = blockIdx.x * 4 + wave; tile < nTiles; tile += gridDim.x * 4) {
    const int n0 = tile << 4;
    int node = n0 + ei; if (node >= NNODES) node = NNODES - 1;
    // concat row: cols 0..63 = x[node], cols 64..191 = agg[node]
    #pragma unroll
    for (int s = 0; s < 3; s++) {
      int cc = hi * 96 + s * 32;
      const float* src = (cc < 64) ? (x + (long long)node * FIN + cc)
                                   : (agg + (long long)node * HID + (cc - 64));
      _Float16* arow = A + ei * KC0 + cc;
      #pragma unroll
      for (int f = 0; f < 32; f += 4) {
        float4 v = *(const float4*)(src + f);
        v4h h4 = { (_Float16)v.x, (_Float16)v.y, (_Float16)v.z, (_Float16)v.w };
        *(v4h*)(arow + f) = h4;
      }
    }

    // ---- layer 0: 16x192 @ 192x128 ----
    {
      v16h a[6];
      #pragma unroll
      for (int ks = 0; ks < 6; ks++) a[ks] = lds_loadA(A, KC0, lane, ks * 32);
      #pragma unroll
      for (int nt = 0; nt < 8; nt++) {
        float bv = sB0[nt * 16 + col0];
        v8f acc = { bv, bv, bv, bv, bv, bv, bv, bv };
        const _Float16* wb = sW0 + nt * 16 * KC0;
        #pragma unroll
        for (int ks = 0; ks < 6; ks++)
          acc = wmma16(a[ks], lds_loadB(wb, KC0, lane, ks * 32), acc);
        #pragma unroll
        for (int j = 0; j < 8; j++) {
          float v = acc[j]; v = v > 0.0f ? v : 0.0f;
          A[(j + hi * 8) * KC0 + nt * 16 + col0] = (_Float16)v;
        }
      }
    }
    // ---- layers 1 & 2: 16x128 @ 128x128 ----
    #pragma unroll 1
    for (int layer = 0; layer < 2; layer++) {
      const _Float16* W = layer ? sW2 : sW1;
      const float*    B = layer ? sB2 : sB1;
      v16h h[4];
      #pragma unroll
      for (int ks = 0; ks < 4; ks++) h[ks] = lds_loadA(A, KC0, lane, ks * 32);
      #pragma unroll
      for (int nt = 0; nt < 8; nt++) {
        float bv = B[nt * 16 + col0];
        v8f acc = { bv, bv, bv, bv, bv, bv, bv, bv };
        const _Float16* wb = W + nt * 16 * HID;
        #pragma unroll
        for (int ks = 0; ks < 4; ks++)
          acc = wmma16(h[ks], lds_loadB(wb, HID, lane, ks * 32), acc);
        #pragma unroll
        for (int j = 0; j < 8; j++) {
          float v = acc[j]; v = v > 0.0f ? v : 0.0f;
          A[(j + hi * 8) * KC0 + nt * 16 + col0] = (_Float16)v;
        }
      }
    }
    // ---- layer 3: 16x128 @ 128x16 (cols 0..2 valid) ----
    {
      v16h h[4];
      #pragma unroll
      for (int ks = 0; ks < 4; ks++) h[ks] = lds_loadA(A, KC0, lane, ks * 32);
      float bv = sB3[col0];
      v8f acc = { bv, bv, bv, bv, bv, bv, bv, bv };
      #pragma unroll
      for (int ks = 0; ks < 4; ks++)
        acc = wmma16(h[ks], lds_loadB(sW3, HID, lane, ks * 32), acc);
      if (col0 < 3) {
        #pragma unroll
        for (int j = 0; j < 8; j++) {
          int nd = n0 + j + hi * 8;
          if (nd < NNODES) out[(long long)nd * 3 + col0] = acc[j];
        }
      }
    }
  }
}

extern "C" void kernel_launch(void* const* d_in, const int* in_sizes, int n_in,
                              void* d_out, int out_size, void* d_ws, size_t ws_size,
                              hipStream_t stream) {
  const float*     x    = (const float*)d_in[0];
  const long long* eidx = (const long long*)d_in[1];
  const float* mW0 = (const float*)d_in[2];
  const float* mb0 = (const float*)d_in[3];
  const float* mW1 = (const float*)d_in[4];
  const float* mb1 = (const float*)d_in[5];
  const float* mW2 = (const float*)d_in[6];
  const float* mb2 = (const float*)d_in[7];
  const float* cW0 = (const float*)d_in[8];
  const float* cb0 = (const float*)d_in[9];
  const float* cW1 = (const float*)d_in[10];
  const float* cb1 = (const float*)d_in[11];
  const float* cW2 = (const float*)d_in[12];
  const float* cb2 = (const float*)d_in[13];
  const float* cW3 = (const float*)d_in[14];
  const float* cb3 = (const float*)d_in[15];
  float* agg = (float*)d_ws;          // N*H f32 = 51.2 MB scratch
  float* out = (float*)d_out;

  zero_kernel<<<2048, 256, 0, stream>>>(agg, (NNODES * HID) / 4);
  edge_msg_kernel<<<1024, 256, 0, stream>>>(x, eidx, mW0, mb0, mW1, mb1, mW2, mb2, agg);
  corrector_kernel<<<1024, 128, 0, stream>>>(x, agg, cW0, cb0, cW1, cb1, cW2, cb2,
                                             cW3, cb3, out);
}
